// simple_model_terms_51462298141020
// MI455X (gfx1250) — matrix operand order repaired and checked by reading, not compiled, not run on hardware
//
#include <hip/hip_runtime.h>
#include <type_traits>

// CDNA5 / gfx1250, wave32.
// Fused: gather(rf, roi) + leaky_relu -> GEMM via error-compensated bf16 WMMA
// (x = hi + lo truncation split; hi*hi + hi*lo + lo*hi, f32 accum) -> * gather(w, roi) -> scatter.
// Double-buffered LDS: stage tile kt+1 while WMMAs consume tile kt (one barrier per tile).

typedef __bf16 bf16_t;
typedef bf16_t v16bf __attribute__((ext_vector_type(16)));
typedef float v8f __attribute__((ext_vector_type(8)));
typedef unsigned int v4u __attribute__((ext_vector_type(4)));

#define NV_TOTAL 20000
#define NV_ROI   8000
#define OUT_CH   960
#define BATCH    8

#define BM 128          // block tile M
#define BN 64           // block tile N (8000 = 125 * 64, exact)
#define KT 32           // K tile (one bf16 WMMA K-depth)
#define KTP (KT + 8)    // padded bf16 row stride (80 B): conflict-free b128 reads

// Pack the high 16 bits (truncated bf16) of two f32 into one u32: {x1.hi16, x0.hi16}.
static __device__ inline unsigned pack_hi16(float x0, float x1) {
    return __builtin_amdgcn_perm(__float_as_uint(x1), __float_as_uint(x0), 0x07060302u);
}
// Truncation split: hi = x masked to bf16, lo = trunc16(x - hi). Two elements -> two packed u32.
static __device__ inline void split_pack2(float x0, float x1, unsigned& hi, unsigned& lo) {
    hi = pack_hi16(x0, x1);
    const float r0 = x0 - __uint_as_float(__float_as_uint(x0) & 0xFFFF0000u);
    const float r1 = x1 - __uint_as_float(__float_as_uint(x1) & 0xFFFF0000u);
    lo = pack_hi16(r0, r1);
}
// leaky_relu(0.01) without compares: max(x,0) + 0.01*min(x,0)
static __device__ inline float leaky01(float x) {
    return fmaf(0.01f, fminf(x, 0.0f), fmaxf(x, 0.0f));
}

// A fragment (16x32 bf16): lane l<16 -> row l, K chunks {kb..kb+7, kb+16..kb+23}, kb=(l>>4)*8
static __device__ inline v16bf load_fragA(const unsigned short* rowbase, int lane) {
    const int kb = (lane >> 4) * 8;
    union { v4u q[2]; v16bf v; } u;
    u.q[0] = *(const v4u*)(rowbase + kb);
    u.q[1] = *(const v4u*)(rowbase + kb + 16);
    return u.v;
}
// B fragment (32x16 bf16): lane l<16 -> col l, K chunks {kb..kb+7, kb+16..kb+23}, kb=(l>>4)*8
static __device__ inline v16bf load_fragB(const unsigned short* colbase, int lane) {
    const int kb = (lane >> 4) * 8;
    union { v4u q[2]; v16bf v; } u;
    u.q[0] = *(const v4u*)(colbase + kb);
    u.q[1] = *(const v4u*)(colbase + kb + 16);
    return u.v;
}

template <int C, int P, int CH_OFF>
__global__ __launch_bounds__(256)
void fused_roi_gemm_bf16x3(const float* __restrict__ fmap,    // [BATCH, C, P]
                           const float* __restrict__ rf,      // [P, NV_TOTAL]
                           const float* __restrict__ w,       // [C, NV_TOTAL]
                           const long long* __restrict__ roi, // [NV_ROI] int64 sorted
                           float* __restrict__ out)           // [BATCH, OUT_CH, NV_ROI]
{
    __shared__ unsigned short AsHi[2][BM][KTP];
    __shared__ unsigned short AsLo[2][BM][KTP];
    __shared__ unsigned short BsHi[2][BN][KTP];   // n-major: Bs[n][k]
    __shared__ unsigned short BsLo[2][BN][KTP];
    __shared__ int roiS[BN];

    const int tid  = threadIdx.x;
    const int lane = tid & 31;
    const int wave = tid >> 5;           // 0..7
    const int wm   = (wave >> 1) * 32;   // wave M offset (0,32,64,96)
    const int wn   = (wave & 1) * 32;    // wave N offset (0,32)
    const int l15  = lane & 15;

    const int mg0 = blockIdx.y * BM;
    const int ng0 = blockIdx.x * BN;

    if (tid < BN) roiS[tid] = (int)roi[ng0 + tid];
    __syncthreads();

    v8f acc[2][2];
    #pragma unroll
    for (int i = 0; i < 2; ++i)
        #pragma unroll
        for (int j = 0; j < 2; ++j)
            acc[i][j] = (v8f){0.f, 0.f, 0.f, 0.f, 0.f, 0.f, 0.f, 0.f};

    // staging geometry (fixed per thread)
    const int arow = tid >> 1;           // 0..127
    const int akk  = (tid & 1) * 16;     // 0 or 16
    const int bn   = tid & 63;           // 0..63
    const int bkb  = (tid >> 6) * 8;     // 0,8,16,24
    const int bcol = roiS[bn];
    const float* fmrow = fmap + (size_t)(mg0 + arow) * P + akk;

    // ---- stage one K tile into buffer `buf` (guarded for the K remainder) ----
    auto stage = [&](int k0, int buf, auto guardc) {
        constexpr bool G = decltype(guardc)::value;
        // A: 16 consecutive k per thread
        {
            const float* src = fmrow + k0;
            unsigned* dHi = (unsigned*)&AsHi[buf][arow][akk];
            unsigned* dLo = (unsigned*)&AsLo[buf][arow][akk];
            #pragma unroll
            for (int j = 0; j < 16; j += 2) {
                float x0, x1;
                if constexpr (G) {
                    x0 = (k0 + akk + j     < P) ? src[j]     : 0.0f;
                    x1 = (k0 + akk + j + 1 < P) ? src[j + 1] : 0.0f;
                } else {
                    x0 = src[j];
                    x1 = src[j + 1];
                }
                unsigned hi, lo;
                split_pack2(x0, x1, hi, lo);
                dHi[j >> 1] = hi;
                dLo[j >> 1] = lo;
            }
        }
        // B: gather + leaky_relu, 8 k per thread
        {
            unsigned* dHi = (unsigned*)&BsHi[buf][bn][bkb];
            unsigned* dLo = (unsigned*)&BsLo[buf][bn][bkb];
            #pragma unroll
            for (int j = 0; j < 8; j += 2) {
                float x0 = 0.0f, x1 = 0.0f;
                if (!G || (k0 + bkb + j < P))
                    x0 = leaky01(rf[(size_t)(k0 + bkb + j) * NV_TOTAL + bcol]);
                if (!G || (k0 + bkb + j + 1 < P))
                    x1 = leaky01(rf[(size_t)(k0 + bkb + j + 1) * NV_TOTAL + bcol]);
                unsigned hi, lo;
                split_pack2(x0, x1, hi, lo);
                dHi[j >> 1] = hi;
                dLo[j >> 1] = lo;
            }
        }
    };

    // ---- fragment loads from buffer `buf`, then 12 compensated WMMAs ----
    auto compute = [&](int buf) {
        v16bf aH[2], aL[2], bH[2], bL[2];
        #pragma unroll
        for (int mt = 0; mt < 2; ++mt) {
            const int row = wm + mt * 16 + l15;
            aH[mt] = load_fragA(&AsHi[buf][row][0], lane);
            aL[mt] = load_fragA(&AsLo[buf][row][0], lane);
        }
        #pragma unroll
        for (int nt = 0; nt < 2; ++nt) {
            const int coln = wn + nt * 16 + l15;
            bH[nt] = load_fragB(&BsHi[buf][coln][0], lane);
            bL[nt] = load_fragB(&BsLo[buf][coln][0], lane);
        }
        #pragma unroll
        for (int mt = 0; mt < 2; ++mt)
            #pragma unroll
            for (int nt = 0; nt < 2; ++nt) {
                v8f c = acc[mt][nt];
                c = __builtin_amdgcn_wmma_f32_16x16x32_bf16(false, aH[mt], false, bL[nt],
                                                            (short)0, c, false, false);
                c = __builtin_amdgcn_wmma_f32_16x16x32_bf16(false, aL[mt], false, bH[nt],
                                                            (short)0, c, false, false);
                c = __builtin_amdgcn_wmma_f32_16x16x32_bf16(false, aH[mt], false, bH[nt],
                                                            (short)0, c, false, false);
                acc[mt][nt] = c;
            }
    };

    constexpr int KFULL = P / KT;                       // full (unguarded) tiles
    constexpr bool HAS_REM = (P % KT) != 0;
    constexpr int NKT = KFULL + (HAS_REM ? 1 : 0);      // >= 1 for all layers (min P = 49)

    // prologue: tile 0 is always a full tile (P >= KT for all layers)
    stage(0, 0, std::false_type{});
    __syncthreads();

    for (int kt = 0; kt < NKT; ++kt) {
        const int cur = kt & 1;
        if (kt + 1 < NKT) {
            const int k1 = (kt + 1) * KT;
            if (HAS_REM && (kt + 1 == NKT - 1))
                stage(k1, cur ^ 1, std::true_type{});
            else
                stage(k1, cur ^ 1, std::false_type{});
            if (kt + 2 < NKT)  // warm L2 for the tile after next
                __builtin_prefetch(&rf[(size_t)((kt + 2) * KT + bkb) * NV_TOTAL + bcol], 0, 1);
        }
        compute(cur);
        __syncthreads();
    }

    // ---- epilogue: scale by gathered w, scatter into concatenated output ----
    // C/D layout: VGPR r, lanes 0-15 -> M=r, N=lane; lanes 16-31 -> M=r+8, N=lane-16.
    const int mhi = (lane >> 4) * 8;
    #pragma unroll
    for (int mt = 0; mt < 2; ++mt) {
        #pragma unroll
        for (int nt = 0; nt < 2; ++nt) {
            const int ncol_l = wn + nt * 16 + l15;
            const int nglob  = ng0 + ncol_l;
            const int ridx   = roiS[ncol_l];
            #pragma unroll
            for (int r = 0; r < 8; ++r) {
                const int mrow = mg0 + wm + mt * 16 + r + mhi;
                const int b = mrow / C;
                const int c = mrow - b * C;
                const float scale = w[(size_t)c * NV_TOTAL + ridx];
                out[(size_t)b * OUT_CH * NV_ROI + (size_t)(CH_OFF + c) * NV_ROI + nglob] =
                    acc[mt][nt][r] * scale;
            }
        }
    }
}

extern "C" void kernel_launch(void* const* d_in, const int* in_sizes, int n_in,
                              void* d_out, int out_size, void* d_ws, size_t ws_size,
                              hipStream_t stream) {
    (void)in_sizes; (void)n_in; (void)out_size; (void)d_ws; (void)ws_size;

    // input order: fmap0, rf0, w0, fmap1, rf1, w1, fmap2, rf2, w2, fmap3, rf3, w3, roi_idx
    const float* fmap0 = (const float*)d_in[0];
    const float* rf0   = (const float*)d_in[1];
    const float* w0    = (const float*)d_in[2];
    const float* fmap1 = (const float*)d_in[3];
    const float* rf1   = (const float*)d_in[4];
    const float* w1    = (const float*)d_in[5];
    const float* fmap2 = (const float*)d_in[6];
    const float* rf2   = (const float*)d_in[7];
    const float* w2    = (const float*)d_in[8];
    const float* fmap3 = (const float*)d_in[9];
    const float* rf3   = (const float*)d_in[10];
    const float* w3    = (const float*)d_in[11];
    const long long* roi = (const long long*)d_in[12];
    float* out = (float*)d_out;

    dim3 blk(256);
    const int nblk = NV_ROI / BN;  // 125

    fused_roi_gemm_bf16x3<64, 3136, 0><<<dim3(nblk, 512 / BM), blk, 0, stream>>>(fmap0, rf0, w0, roi, out);
    fused_roi_gemm_bf16x3<128, 784, 64><<<dim3(nblk, 1024 / BM), blk, 0, stream>>>(fmap1, rf1, w1, roi, out);
    fused_roi_gemm_bf16x3<256, 196, 192><<<dim3(nblk, 2048 / BM), blk, 0, stream>>>(fmap2, rf2, w2, roi, out);
    fused_roi_gemm_bf16x3<512, 49, 448><<<dim3(nblk, 4096 / BM), blk, 0, stream>>>(fmap3, rf3, w3, roi, out);
}
